// NonParametricProtoCritic_63677185130825
// MI455X (gfx1250) — matrix operand order repaired and checked by reading, not compiled, not run on hardware
//
#include <hip/hip_runtime.h>
#include <hip/hip_bf16.h>
#include <math.h>

#define B_ROWS   4096
#define OBS_DIMC 1024
#define ACT_DIMC 6
#define KEY_DIMC 512
#define NUM_KEYS 16384
#define TOP_K    32
#define INV_TEMP 10.0f
#define NEG_INF  -3.0e38f

typedef __attribute__((ext_vector_type(16))) __bf16 v16bf;
typedef __attribute__((ext_vector_type(8)))  float  v8f;

union FragU {
  v16bf v;
  uint4 q[2];
};

__device__ __forceinline__ unsigned short f32_to_bf16(float f) {
  unsigned int u = __float_as_uint(f);
  u += 0x7fffu + ((u >> 16) & 1u);   // round-to-nearest-even
  return (unsigned short)(u >> 16);
}

// A fragment (16x32 bf16, M x K) from an LDS tile laid out row-major [16][ldk].
__device__ __forceinline__ v16bf load_frag_a_lds(const unsigned short* smemA, int ldk, int kb) {
  const int lane = threadIdx.x & 31;
  const int m    = lane & 15;
  const int half = lane >> 4;
  const unsigned short* p = smemA + m * ldk + kb + half * 8;
  FragU f;
  f.q[0] = *(const uint4*)(p);
  f.q[1] = *(const uint4*)(p + 16);
  return f.v;
}

// B fragment (32x16 bf16, K x N) from global weights stored row-major [N][ldb].
__device__ __forceinline__ v16bf load_frag_b_glb(const unsigned short* Bm, int ldb, int n0, int kb) {
  const int lane = threadIdx.x & 31;
  const int n    = lane & 15;
  const int half = lane >> 4;
  const unsigned short* p = Bm + (size_t)(n0 + n) * ldb + kb + half * 8;
  FragU f;
  f.q[0] = *(const uint4*)(p);
  f.q[1] = *(const uint4*)(p + 16);
  return f.v;
}

// ---------------------------------------------------------------------------
// elementwise f32 -> bf16 (independent src/dst row strides)
// ---------------------------------------------------------------------------
__global__ void cvt_f32_bf16_strided(const float* __restrict__ src, unsigned short* __restrict__ dst,
                                     int rows, int cols, int sld, int dld) {
  long long i = (long long)blockIdx.x * blockDim.x + threadIdx.x;
  long long total = (long long)rows * cols;
  if (i >= total) return;
  int r = (int)(i / cols), c = (int)(i % cols);
  dst[(size_t)r * dld + c] = f32_to_bf16(src[(size_t)r * sld + c]);
}

// ---------------------------------------------------------------------------
// row-wise L2 normalize (f32 in, bf16 out), one row per 128-thread block
// ---------------------------------------------------------------------------
__global__ __launch_bounds__(128) void l2norm_rows_bf16(const float* __restrict__ in,
                                                        unsigned short* __restrict__ out,
                                                        int cols) {
  const int row = blockIdx.x;
  const float* p = in + (size_t)row * cols;
  float ss = 0.0f;
  for (int c = threadIdx.x; c < cols; c += blockDim.x) {
    float x = p[c];
    ss = fmaf(x, x, ss);
  }
  for (int off = 16; off > 0; off >>= 1) ss += __shfl_xor(ss, off, 32);
  __shared__ float wsum[4];
  const int lane = threadIdx.x & 31, wv = threadIdx.x >> 5;
  if (lane == 0) wsum[wv] = ss;
  __syncthreads();
  const float tot = wsum[0] + wsum[1] + wsum[2] + wsum[3];
  const float inv = 1.0f / fmaxf(sqrtf(tot), 1e-12f);
  unsigned short* q = out + (size_t)row * cols;
  for (int c = threadIdx.x; c < cols; c += blockDim.x)
    q[c] = f32_to_bf16(p[c] * inv);
}

// ---------------------------------------------------------------------------
// WMMA GEMM, M-tile 32 (2 accumulators -> each B fragment feeds 2 WMMAs):
// C[M,N] = A[M,K] (bf16, lda==K) @ W[N,K]^T (bf16, ldb) + bias
// EPI 0: store f32.   EPI 1: += action@W1b^T, ReLU, store bf16.
// grid.x = M/32, block = 128 (4 waves).
// ---------------------------------------------------------------------------
template <int EPI>
__global__ __launch_bounds__(128) void gemm32_bf16(
    const unsigned short* __restrict__ A, const unsigned short* __restrict__ Bm,
    const float* __restrict__ bias, float* __restrict__ Cf, unsigned short* __restrict__ Cbf,
    int N, int K, int ldb,
    const float* __restrict__ act, const float* __restrict__ w1b, int w1b_ld) {
  __shared__ unsigned short smemA[32 * 1024];   // up to K=1024 (64 KB)
  const int m0 = blockIdx.x * 32;

  {  // stage A tile (contiguous: rows m0..m0+31, full K, lda == K)
    const uint4* src = (const uint4*)(A + (size_t)m0 * K);
    uint4* dst = (uint4*)smemA;
    const int nvec = (32 * K) / 8;
    for (int i = threadIdx.x; i < nvec; i += blockDim.x) dst[i] = src[i];
  }
  __syncthreads();

  const int wave  = threadIdx.x >> 5;
  const int lane  = threadIdx.x & 31;
  const int col   = lane & 15;
  const int halfI = lane >> 4;

  for (int nt = wave; nt < N / 16; nt += 4) {
    const int n0 = nt * 16;
    v8f acc0 = {0.f, 0.f, 0.f, 0.f, 0.f, 0.f, 0.f, 0.f};
    v8f acc1 = {0.f, 0.f, 0.f, 0.f, 0.f, 0.f, 0.f, 0.f};
    for (int kb = 0; kb < K; kb += 32) {
      v16bf b  = load_frag_b_glb(Bm, ldb, n0, kb);
      v16bf a0 = load_frag_a_lds(smemA, K, kb);
      v16bf a1 = load_frag_a_lds(smemA + 16 * K, K, kb);
      acc0 = __builtin_amdgcn_wmma_f32_16x16x32_bf16(false, a0, false, b, (short)0, acc0, false, false);
      acc1 = __builtin_amdgcn_wmma_f32_16x16x32_bf16(false, a1, false, b, (short)0, acc1, false, false);
    }
    const int n = n0 + col;
    const float bv = bias[n];
    auto epilogue = [&](v8f acc, int msub) {
#pragma unroll
      for (int v = 0; v < 8; ++v) {
        const int m = m0 + msub + v + 8 * halfI;
        float c = acc[v] + bv;
        if (EPI == 1) {
#pragma unroll
          for (int j = 0; j < ACT_DIMC; ++j)
            c = fmaf(act[m * ACT_DIMC + j], w1b[n * w1b_ld + j], c);
          c = fmaxf(c, 0.0f);
          Cbf[(size_t)m * N + n] = f32_to_bf16(c);
        } else {
          Cf[(size_t)m * N + n] = c;
        }
      }
    };
    epilogue(acc0, 0);
    epilogue(acc1, 16);
  }
}

// ---------------------------------------------------------------------------
// Fused similarity GEMM + exact per-row top-32 for one key half.
// grid = (B/32, 2): 32 query rows per block, gridDim.y splits the 16384 keys.
// Block = 256 (8 waves). Each wave streams 16-key chunks over its half with
// 2 accumulators (rows 0-15 / 16-31), so each B fragment feeds 2 WMMAs.
// Lane L keeps the exact top-32 of row L over its disjoint key subset in a
// register-resident sorted list; 8 lane-lists/row are merged by 32 shuffle
// arg-max extractions, emitting a sorted top-32 (sim,val) list per row/half.
// ---------------------------------------------------------------------------
__global__ __launch_bounds__(256) void sim_topk_kernel(
    const unsigned short* __restrict__ qn, const unsigned short* __restrict__ kn,
    const float* __restrict__ values, float* __restrict__ partS, float* __restrict__ partV) {
  __shared__ unsigned char smemRaw[65536];
  unsigned short* smemQ  = (unsigned short*)smemRaw;        // 32 KB (phase 1)
  float*          smemSm = (float*)(smemRaw + 32768);       // 16 KB (phase 1): 8 waves * 32x16

  const int m0   = blockIdx.x * 32;
  const int gy   = blockIdx.y;                              // key half
  const int wave = threadIdx.x >> 5;
  const int lane = threadIdx.x & 31;

  {  // stage qn tile [32 x 512] bf16 (contiguous)
    const uint4* src = (const uint4*)(qn + (size_t)m0 * KEY_DIMC);
    uint4* dst = (uint4*)smemQ;
    for (int i = threadIdx.x; i < (32 * KEY_DIMC) / 8; i += blockDim.x) dst[i] = src[i];
  }
  __syncthreads();

  float ts[TOP_K], tv[TOP_K];
#pragma unroll
  for (int i = 0; i < TOP_K; ++i) { ts[i] = NEG_INF; tv[i] = 0.0f; }

  float* stile = smemSm + wave * 512;   // this wave's private 32x16 staging

  const int ncEnd = (gy + 1) * (NUM_KEYS / 32);             // chunks of 16 keys
  for (int nc = gy * (NUM_KEYS / 32) + wave; nc < ncEnd; nc += 8) {
    const int n0 = nc * 16;
    if (n0 + 128 < NUM_KEYS)
      __builtin_prefetch(kn + (size_t)(n0 + 128) * KEY_DIMC, 0, 1);

    v8f acc0 = {0.f, 0.f, 0.f, 0.f, 0.f, 0.f, 0.f, 0.f};
    v8f acc1 = {0.f, 0.f, 0.f, 0.f, 0.f, 0.f, 0.f, 0.f};
    for (int kb = 0; kb < KEY_DIMC; kb += 32) {
      v16bf b  = load_frag_b_glb(kn, KEY_DIMC, n0, kb);
      v16bf a0 = load_frag_a_lds(smemQ, KEY_DIMC, kb);
      v16bf a1 = load_frag_a_lds(smemQ + 16 * KEY_DIMC, KEY_DIMC, kb);
      acc0 = __builtin_amdgcn_wmma_f32_16x16x32_bf16(false, a0, false, b, (short)0, acc0, false, false);
      acc1 = __builtin_amdgcn_wmma_f32_16x16x32_bf16(false, a1, false, b, (short)0, acc1, false, false);
    }
    {  // spill both 16x16 tiles -> 32x16 staging (rows m-major)
      const int colw = lane & 15, halfI = lane >> 4;
#pragma unroll
      for (int v = 0; v < 8; ++v) {
        stile[(v + 8 * halfI) * 16 + colw]        = acc0[v];
        stile[(16 + v + 8 * halfI) * 16 + colw]   = acc1[v];
      }
    }
    // consume: lane L owns row L, all 16 cols (in-order LDS within the wave)
    const float* myrow = stile + lane * 16;
#pragma unroll
    for (int j = 0; j < 16; ++j) {
      float s = myrow[j] * INV_TEMP;
      if (s > ts[0]) {                              // rare after warm-up
        float cur  = s;
        float curv = values[n0 + j];
#pragma unroll
        for (int i = 0; i < TOP_K; ++i) {           // sorted insert, evict min
          const float nxt  = (i < TOP_K - 1) ? ts[i + 1] : 3.4e38f;
          const float nxtv = (i < TOP_K - 1) ? tv[i + 1] : 0.0f;
          if (nxt < cur) { ts[i] = nxt; tv[i] = nxtv; }
          else           { ts[i] = cur; tv[i] = curv; break; }
        }
      }
    }
  }

  __syncthreads();
  // dump candidate lists: row-major [32 rows][8 waves][32]  (aliases phase-1 LDS)
  float* candS = (float*)smemRaw;            // 32 KB
  float* candV = (float*)(smemRaw + 32768);  // 32 KB
  {
    float* ps = candS + (lane * 8 + wave) * TOP_K;
    float* pv = candV + (lane * 8 + wave) * TOP_K;
#pragma unroll
    for (int i = 0; i < TOP_K; ++i) { ps[i] = ts[i]; pv[i] = tv[i]; }
  }
  __syncthreads();

  // per-row exact top-32 of this key half, sorted descending: wave -> 4 rows
  for (int rr = wave * 4; rr < wave * 4 + 4; ++rr) {
    float* rs = candS + rr * 256;
    const float* rv = candV + rr * 256;
    const int base = lane * 8;
    float* oS = partS + ((size_t)gy * B_ROWS + m0 + rr) * TOP_K;
    float* oV = partV + ((size_t)gy * B_ROWS + m0 + rr) * TOP_K;
    for (int it = 0; it < TOP_K; ++it) {
      float bm = NEG_INF;
      int bj = 0;
#pragma unroll
      for (int j = 0; j < 8; ++j) {
        const float x = rs[base + j];
        if (x > bm) { bm = x; bj = j; }
      }
      int bidx = base + bj;
      for (int off = 16; off > 0; off >>= 1) {
        const float om = __shfl_xor(bm, off, 32);
        const int   oi = __shfl_xor(bidx, off, 32);
        if (om > bm) { bm = om; bidx = oi; }
      }
      if (lane == 0) {
        oS[it] = bm;
        oV[it] = rv[bidx];
        rs[bidx] = NEG_INF;            // in-order LDS: visible next iteration
      }
    }
  }
}

// ---------------------------------------------------------------------------
// Final 2-way sorted merge of the per-half top-32 lists + softmax + value sum.
// One thread per query row.
// ---------------------------------------------------------------------------
__global__ __launch_bounds__(256) void final_merge_kernel(
    const float* __restrict__ partS, const float* __restrict__ partV, float* __restrict__ out) {
  const int m = blockIdx.x * blockDim.x + threadIdx.x;
  if (m >= B_ROWS) return;
  const float* sA = partS + (size_t)m * TOP_K;
  const float* vA = partV + (size_t)m * TOP_K;
  const float* sB = partS + ((size_t)B_ROWS + m) * TOP_K;
  const float* vB = partV + ((size_t)B_ROWS + m) * TOP_K;
  const float smax = fmaxf(sA[0], sB[0]);   // both lists sorted descending
  int ia = 0, ib = 0;
  float num = 0.0f, den = 0.0f;
  for (int it = 0; it < TOP_K; ++it) {
    const float a = (ia < TOP_K) ? sA[ia] : NEG_INF;
    const float b = (ib < TOP_K) ? sB[ib] : NEG_INF;
    const bool ta = (a >= b);
    const float s = ta ? a : b;
    const float v = ta ? vA[ia] : vB[ib];
    if (ta) ++ia; else ++ib;
    const float e = __expf(s - smax);
    num = fmaf(e, v, num);
    den += e;
  }
  const float q = num / den;
  out[m] = q;
  out[B_ROWS + m] = q;                       // reference returns (q, q)
}

// ---------------------------------------------------------------------------
extern "C" void kernel_launch(void* const* d_in, const int* in_sizes, int n_in,
                              void* d_out, int out_size, void* d_ws, size_t ws_size,
                              hipStream_t stream) {
  (void)in_sizes; (void)n_in; (void)out_size; (void)ws_size;
  const float* obs    = (const float*)d_in[0];
  const float* action = (const float*)d_in[1];
  const float* W_pred = (const float*)d_in[2];
  const float* b_pred = (const float*)d_in[3];
  const float* W1     = (const float*)d_in[4];
  const float* b1     = (const float*)d_in[5];
  const float* W2     = (const float*)d_in[6];
  const float* b2     = (const float*)d_in[7];
  const float* keys   = (const float*)d_in[8];
  const float* values = (const float*)d_in[9];
  float* out = (float*)d_out;

  char* ws = (char*)d_ws;
  size_t off = 0;
  auto carve = [&](size_t bytes) -> char* {
    char* p = ws + off;
    off += (bytes + 255) & ~(size_t)255;
    return p;
  };
  unsigned short* obs_bf = (unsigned short*)carve((size_t)B_ROWS * OBS_DIMC * 2);
  unsigned short* wp_bf  = (unsigned short*)carve((size_t)KEY_DIMC * OBS_DIMC * 2);
  unsigned short* w1_bf  = (unsigned short*)carve((size_t)KEY_DIMC * KEY_DIMC * 2);
  unsigned short* w2_bf  = (unsigned short*)carve((size_t)KEY_DIMC * KEY_DIMC * 2);
  unsigned short* kn_bf  = (unsigned short*)carve((size_t)NUM_KEYS * KEY_DIMC * 2);
  float*          pre_f  = (float*)carve((size_t)B_ROWS * KEY_DIMC * 4);
  unsigned short* in_bf  = (unsigned short*)carve((size_t)B_ROWS * KEY_DIMC * 2);
  unsigned short* h1_bf  = (unsigned short*)carve((size_t)B_ROWS * KEY_DIMC * 2);
  float*          h2_f   = (float*)carve((size_t)B_ROWS * KEY_DIMC * 4);
  unsigned short* qn_bf  = (unsigned short*)carve((size_t)B_ROWS * KEY_DIMC * 2);
  float*          partS  = (float*)carve((size_t)2 * B_ROWS * TOP_K * 4);
  float*          partV  = (float*)carve((size_t)2 * B_ROWS * TOP_K * 4);

  const dim3 blk(256);
  auto g1 = [](long long n) { return dim3((unsigned)((n + 255) / 256)); };

  // precision staging (bf16 WMMA operands, f32 accumulate everywhere)
  cvt_f32_bf16_strided<<<g1((long long)B_ROWS * OBS_DIMC), blk, 0, stream>>>(obs, obs_bf, B_ROWS, OBS_DIMC, OBS_DIMC, OBS_DIMC);
  cvt_f32_bf16_strided<<<g1((long long)KEY_DIMC * OBS_DIMC), blk, 0, stream>>>(W_pred, wp_bf, KEY_DIMC, OBS_DIMC, OBS_DIMC, OBS_DIMC);
  cvt_f32_bf16_strided<<<g1((long long)KEY_DIMC * KEY_DIMC), blk, 0, stream>>>(W1, w1_bf, KEY_DIMC, KEY_DIMC, KEY_DIMC + ACT_DIMC, KEY_DIMC);
  cvt_f32_bf16_strided<<<g1((long long)KEY_DIMC * KEY_DIMC), blk, 0, stream>>>(W2, w2_bf, KEY_DIMC, KEY_DIMC, KEY_DIMC, KEY_DIMC);
  l2norm_rows_bf16<<<dim3(NUM_KEYS), dim3(128), 0, stream>>>(keys, kn_bf, KEY_DIMC);

  // predictor + normalize
  gemm32_bf16<0><<<dim3(B_ROWS / 32), dim3(128), 0, stream>>>(
      obs_bf, wp_bf, b_pred, pre_f, nullptr, KEY_DIMC, OBS_DIMC, OBS_DIMC, nullptr, nullptr, 0);
  l2norm_rows_bf16<<<dim3(B_ROWS), dim3(128), 0, stream>>>(pre_f, in_bf, KEY_DIMC);

  // trunk: relu(inpt@W1a^T + action@W1b^T + b1) @ W2^T + b2
  gemm32_bf16<1><<<dim3(B_ROWS / 32), dim3(128), 0, stream>>>(
      in_bf, w1_bf, b1, nullptr, h1_bf, KEY_DIMC, KEY_DIMC, KEY_DIMC,
      action, W1 + KEY_DIMC, KEY_DIMC + ACT_DIMC);
  gemm32_bf16<0><<<dim3(B_ROWS / 32), dim3(128), 0, stream>>>(
      h1_bf, w2_bf, b2, h2_f, nullptr, KEY_DIMC, KEY_DIMC, KEY_DIMC, nullptr, nullptr, 0);
  l2norm_rows_bf16<<<dim3(B_ROWS), dim3(128), 0, stream>>>(h2_f, qn_bf, KEY_DIMC);

  // fused cosine-sim GEMM + exact top-32 per key half, then 2-way merge+softmax
  sim_topk_kernel<<<dim3(B_ROWS / 32, 2), dim3(256), 0, stream>>>(qn_bf, kn_bf, values, partS, partV);
  final_merge_kernel<<<g1(B_ROWS), blk, 0, stream>>>(partS, partV, out);
}
